// Head_12154757447789
// MI455X (gfx1250) — matrix operand order repaired and checked by reading, not compiled, not run on hardware
//
#include <hip/hip_runtime.h>

typedef _Float16 h16 __attribute__((ext_vector_type(16)));
typedef _Float16 h4  __attribute__((ext_vector_type(4)));
typedef float    f8  __attribute__((ext_vector_type(8)));

union Frag {
    h16   v;
    uint4 u[2];
};

static constexpr int BB = 8;      // batch
static constexpr int TT = 4096;   // sequence
static constexpr int CC = 384;    // embed
static constexpr int HH = 64;     // head size

// ---- CDNA5 async global->LDS copy (ASYNCcnt path), with graceful fallback ----
#if defined(__HIP_DEVICE_COMPILE__) && __has_builtin(__builtin_amdgcn_global_load_async_to_lds_b128)
#define HAS_ASYNC 1
typedef int v4i __attribute__((vector_size(4 * sizeof(int))));
typedef __attribute__((address_space(1))) v4i gv4i;   // global (printed as __device__)
typedef __attribute__((address_space(3))) v4i lv4i;   // LDS
#endif

__device__ __forceinline__ void copy16_g2l(const _Float16* g, _Float16* l) {
#ifdef HAS_ASYNC
    __builtin_amdgcn_global_load_async_to_lds_b128(
        (gv4i*)(v4i*)(void*)const_cast<_Float16*>(g),
        (lv4i*)(v4i*)(void*)l, 0, 0);
#else
    *(uint4*)l = *(const uint4*)g;
#endif
}

#if defined(__HIP_DEVICE_COMPILE__) && __has_builtin(__builtin_amdgcn_s_wait_asynccnt)
#define WAIT_ASYNC(n) __builtin_amdgcn_s_wait_asynccnt(n)
#elif defined(HAS_ASYNC)
#define WAIT_ASYNC(n) asm volatile("s_wait_asynccnt " #n ::: "memory")
#else
#define WAIT_ASYNC(n)
#endif

// ---------------------------------------------------------------- cvt kernels
__global__ void cvt_x(const float* __restrict__ x, _Float16* __restrict__ xh, int n4) {
    int i = blockIdx.x * blockDim.x + threadIdx.x;
    if (i >= n4) return;
    float4 f = ((const float4*)x)[i];
    h4 h = { (_Float16)f.x, (_Float16)f.y, (_Float16)f.z, (_Float16)f.w };
    ((h4*)xh)[i] = h;
}

// W (C,H) fp32 -> Wt (H,C) fp16, 3 matrices concatenated
__global__ void cvt_w(const float* __restrict__ Wq, const float* __restrict__ Wk,
                      const float* __restrict__ Wv, _Float16* __restrict__ wt) {
    int i = blockIdx.x * blockDim.x + threadIdx.x;
    if (i >= 3 * CC * HH) return;
    int mat = i / (CC * HH);
    int rem = i - mat * (CC * HH);
    int r = rem / HH;          // row in C
    int c = rem - r * HH;      // col in H
    const float* W = (mat == 0) ? Wq : (mat == 1) ? Wk : Wv;
    wt[(size_t)mat * CC * HH + (size_t)c * CC + r] = (_Float16)W[(size_t)r * HH + c];
}

// ---------------------------------------------------------------- QKV projection
// grid.x = (B*T)/64 blocks, 384 threads = 12 waves.
// wave = 4*mat + row_tile : each wave computes a 16x64 tile of Q, K, or V.
// Q scaled by 1/sqrt(64); V written TRANSPOSED per batch: vt[b][h][t].
__global__ __launch_bounds__(384) void qkv_proj(const _Float16* __restrict__ xh,
                                                const _Float16* __restrict__ wt,
                                                _Float16* __restrict__ qh,
                                                _Float16* __restrict__ kh,
                                                _Float16* __restrict__ vt) {
    const int lane = threadIdx.x & 31;
    const int wave = threadIdx.x >> 5;      // 0..11
    const int rt   = wave & 3;
    const int mat  = wave >> 2;             // 0=Q 1=K 2=V
    const int row0 = blockIdx.x * 64 + rt * 16;   // flattened b*T + t (64 | T)
    const int n_   = lane & 15;
    const int hi   = lane >> 4;             // 0 or 1

    const _Float16* wm = wt + (size_t)mat * CC * HH;
    f8 acc[4] = {};

    for (int kc = 0; kc < CC; kc += 32) {
        Frag a;
        const _Float16* ap = xh + (size_t)(row0 + n_) * CC + kc + hi * 8;
        a.u[0] = *(const uint4*)ap;
        a.u[1] = *(const uint4*)(ap + 16);
#pragma unroll
        for (int nt = 0; nt < 4; ++nt) {
            Frag b;
            const _Float16* bp = wm + (size_t)(nt * 16 + n_) * CC + kc + hi * 8;
            b.u[0] = *(const uint4*)bp;
            b.u[1] = *(const uint4*)(bp + 16);
            acc[nt] = __builtin_amdgcn_wmma_f32_16x16x32_f16(
                false, a.v, false, b.v, (short)0, acc[nt], false, false);
        }
    }

    if (mat == 2) {           // V: store transposed (b, h, t)
        const int b  = row0 >> 12;          // /TT
        const int t0 = row0 & (TT - 1);
#pragma unroll
        for (int nt = 0; nt < 4; ++nt)
#pragma unroll
            for (int r = 0; r < 8; ++r) {
                int m = r + 8 * hi;
                vt[((size_t)b * HH + nt * 16 + n_) * TT + t0 + m] = (_Float16)acc[nt][r];
            }
    } else {                  // Q (scaled) / K: row-major (b*T, h)
        _Float16* out = (mat == 0) ? qh : kh;
        const float scale = (mat == 0) ? 0.125f : 1.0f;
#pragma unroll
        for (int nt = 0; nt < 4; ++nt)
#pragma unroll
            for (int r = 0; r < 8; ++r) {
                int m = r + 8 * hi;
                out[(size_t)(row0 + m) * HH + nt * 16 + n_] = (_Float16)(acc[nt][r] * scale);
            }
    }
}

// ---------------------------------------------------------------- flash attention
// grid = (T/128, B), 256 threads = 8 waves; wave owns 16 query rows, full H=64.
// Double-buffered async K/V tiles in LDS; online softmax; all GEMMs via WMMA.
__global__ __launch_bounds__(256) void flash_attn(const _Float16* __restrict__ qh,
                                                  const _Float16* __restrict__ kh,
                                                  const _Float16* __restrict__ vt,
                                                  float* __restrict__ out) {
    __shared__ alignas(16) _Float16 Kbuf[2][32 * 64];   // [key][head] row-major
    __shared__ alignas(16) _Float16 Vbuf[2][64 * 32];   // [head][key] (pre-transposed)
    __shared__ alignas(16) _Float16 Plds[8][16 * 32];   // per-wave P tile [m][k]

    const int lane = threadIdx.x & 31;
    const int wave = threadIdx.x >> 5;       // 0..7
    const int b    = blockIdx.y;
    const int q0   = blockIdx.x * 128;
    const int qw   = q0 + wave * 16;         // wave's first query row
    const int n_   = lane & 15;
    const int hi   = lane >> 4;

    // per-thread staging addresses (chunk-invariant parts)
    const int t_     = threadIdx.x;
    const int vrow   = t_ >> 2;              // head row 0..63
    const int vpart  = t_ & 3;               // 16B part within 64B row
    const _Float16* kbase = kh + ((size_t)b * TT) * HH + (size_t)t_ * 8;
    const _Float16* vbase = vt + ((size_t)b * HH + vrow) * TT + vpart * 8;

    // Q A-fragments for head-chunks 0,1 (kept in registers for whole loop)
    Frag qa[2];
    {
        const _Float16* qp = qh + ((size_t)b * TT + qw + n_) * HH + hi * 8;
        qa[0].u[0] = *(const uint4*)(qp);
        qa[0].u[1] = *(const uint4*)(qp + 16);
        qa[1].u[0] = *(const uint4*)(qp + 32);
        qa[1].u[1] = *(const uint4*)(qp + 48);
    }

    f8 acc[4] = {};
    float mrow[8], lrow[8];
#pragma unroll
    for (int r = 0; r < 8; ++r) { mrow[r] = -1e30f; lrow[r] = 0.0f; }

    const int nc = q0 / 32 + 4;              // causal: chunks with key0 <= q0+127

    // stage chunk 0 into buffer 0 (2 async 16B copies per thread)
    copy16_g2l(kbase, &Kbuf[0][t_ * 8]);
    copy16_g2l(vbase, &Vbuf[0][vrow * 32 + vpart * 8]);

    for (int c = 0; c < nc; ++c) {
        const int buf = c & 1;
        if (c + 1 < nc) {
            __syncthreads();   // all waves done reading buf^1 (iteration c-1)
            copy16_g2l(kbase + (size_t)(c + 1) * 32 * HH, &Kbuf[buf ^ 1][t_ * 8]);
            copy16_g2l(vbase + (size_t)(c + 1) * 32,      &Vbuf[buf ^ 1][vrow * 32 + vpart * 8]);
            WAIT_ASYNC(2);     // chunk c's 2 copies (issued earlier, in-order) are done
        } else {
            WAIT_ASYNC(0);
        }
        __syncthreads();       // chunk c tiles staged by all threads
        const _Float16* Kt = &Kbuf[buf][0];
        const _Float16* Vt = &Vbuf[buf][0];

        // -------- S = Q * K^T  (two 16x16 key tiles)
        f8 s[2] = {};
#pragma unroll
        for (int st = 0; st < 2; ++st)
#pragma unroll
            for (int hc = 0; hc < 2; ++hc) {
                Frag bf;
                const _Float16* bp = Kt + (st * 16 + n_) * 64 + hc * 32 + hi * 8;
                bf.u[0] = *(const uint4*)bp;
                bf.u[1] = *(const uint4*)(bp + 16);
                s[st] = __builtin_amdgcn_wmma_f32_16x16x32_f16(
                    false, qa[hc].v, false, bf.v, (short)0, s[st], false, false);
            }

        // -------- causal mask (kj > qi)
#pragma unroll
        for (int r = 0; r < 8; ++r) {
            const int qi = qw + r + 8 * hi;
            const int k0 = c * 32 + n_;
            if (k0 > qi)      s[0][r] = -1e30f;
            if (k0 + 16 > qi) s[1][r] = -1e30f;
        }

        // -------- online softmax (row = r + 8*hi, spread over 16 lanes)
        float alpha[8];
#pragma unroll
        for (int r = 0; r < 8; ++r) {
            float mx = fmaxf(s[0][r], s[1][r]);
#pragma unroll
            for (int d = 1; d < 16; d <<= 1) mx = fmaxf(mx, __shfl_xor(mx, d, 32));
            const float mnew = fmaxf(mrow[r], mx);
            const float p0 = __expf(s[0][r] - mnew);
            const float p1 = __expf(s[1][r] - mnew);
            s[0][r] = p0; s[1][r] = p1;
            float rs = p0 + p1;
#pragma unroll
            for (int d = 1; d < 16; d <<= 1) rs += __shfl_xor(rs, d, 32);
            alpha[r] = __expf(mrow[r] - mnew);
            lrow[r]  = lrow[r] * alpha[r] + rs;
            mrow[r]  = mnew;
        }
#pragma unroll
        for (int nt = 0; nt < 4; ++nt)
#pragma unroll
            for (int r = 0; r < 8; ++r) acc[nt][r] *= alpha[r];

        // -------- reshape P: C-layout -> LDS (row-major 16x32) -> A-layout
        _Float16* pw = &Plds[wave][0];
#pragma unroll
        for (int st = 0; st < 2; ++st)
#pragma unroll
            for (int r = 0; r < 8; ++r)
                pw[(r + 8 * hi) * 32 + st * 16 + n_] = (_Float16)s[st][r];
        asm volatile("s_wait_dscnt 0x0" ::: "memory");   // same-wave LDS turnaround

        Frag pa;
        const _Float16* pp = pw + n_ * 32 + hi * 8;
        pa.u[0] = *(const uint4*)pp;
        pa.u[1] = *(const uint4*)(pp + 16);

        // -------- O += P * V
#pragma unroll
        for (int nt = 0; nt < 4; ++nt) {
            Frag bf;
            const _Float16* bp = Vt + (nt * 16 + n_) * 32 + hi * 8;
            bf.u[0] = *(const uint4*)bp;
            bf.u[1] = *(const uint4*)(bp + 16);
            acc[nt] = __builtin_amdgcn_wmma_f32_16x16x32_f16(
                false, pa.v, false, bf.v, (short)0, acc[nt], false, false);
        }
    }

    // -------- epilogue: divide by row sum, store fp32
#pragma unroll
    for (int nt = 0; nt < 4; ++nt)
#pragma unroll
        for (int r = 0; r < 8; ++r) {
            const int qi = qw + r + 8 * hi;
            out[((size_t)b * TT + qi) * HH + nt * 16 + n_] = acc[nt][r] / lrow[r];
        }
}

// ---------------------------------------------------------------- launcher
extern "C" void kernel_launch(void* const* d_in, const int* in_sizes, int n_in,
                              void* d_out, int out_size, void* d_ws, size_t ws_size,
                              hipStream_t stream) {
    (void)in_sizes; (void)n_in; (void)out_size; (void)ws_size;
    const float* x  = (const float*)d_in[0];
    const float* Wq = (const float*)d_in[1];
    const float* Wk = (const float*)d_in[2];
    const float* Wv = (const float*)d_in[3];
    float* out = (float*)d_out;

    // workspace layout (fp16)
    _Float16* xh = (_Float16*)d_ws;                       // B*T*C
    _Float16* wt = xh + (size_t)BB * TT * CC;             // 3*C*H (transposed)
    _Float16* qh = wt + (size_t)3 * CC * HH;              // B*T*H
    _Float16* kh = qh + (size_t)BB * TT * HH;             // B*T*H
    _Float16* vt = kh + (size_t)BB * TT * HH;             // B*H*T (transposed)

    const int n4 = BB * TT * CC / 4;
    cvt_x<<<(n4 + 255) / 256, 256, 0, stream>>>(x, xh, n4);
    cvt_w<<<(3 * CC * HH + 255) / 256, 256, 0, stream>>>(Wq, Wk, Wv, wt);
    qkv_proj<<<(BB * TT) / 64, 384, 0, stream>>>(xh, wt, qh, kh, vt);
    flash_attn<<<dim3(TT / 128, BB), 256, 0, stream>>>(qh, kh, vt, out);
}
